// JointModel_56178172231981
// MI455X (gfx1250) — matrix operand order repaired and checked by reading, not compiled, not run on hardware
//
#include <hip/hip_runtime.h>
#include <hip/hip_fp16.h>

typedef __attribute__((ext_vector_type(16))) _Float16 v16h;
typedef __attribute__((ext_vector_type(8)))  _Float16 v8h;
typedef __attribute__((ext_vector_type(8)))  float    v8f;

#define H_     768
#define NCLS   7
#define NREL   4
#define MAXL   256
#define BATCH  8
#define SEQ    512
#define NEGINF (-1e9f)

// ---------------- block reduction helpers (256 threads, wave32-safe) ----------------
__device__ __forceinline__ float blockReduceSum256(float v, float* red) {
    int t = threadIdx.x;
    red[t] = v; __syncthreads();
    for (int o = 128; o > 0; o >>= 1) {
        if (t < o) red[t] += red[t + o];
        __syncthreads();
    }
    float r = red[0]; __syncthreads();
    return r;
}
__device__ __forceinline__ float blockReduceMax256(float v, float* red) {
    int t = threadIdx.x;
    red[t] = v; __syncthreads();
    for (int o = 128; o > 0; o >>= 1) {
        if (t < o) red[t] = fmaxf(red[t], red[t + o]);
        __syncthreads();
    }
    float r = red[0]; __syncthreads();
    return r;
}

// ---------------- WMMA fragment helpers (16-bit A 16x32 / B 32x16; contiguous k = 16*hi + s in both) ----------
// A fragment: lane&15 = row M; lane>>4 selects K-halves {0..15} vs {16..31}
__device__ __forceinline__ v16h load_a_frag(const _Float16* __restrict__ rowPtr, int hi) {
    v8h a0 = *(const v8h*)(rowPtr + 16 * hi);       // K = 16*hi + 0..7
    v8h a1 = *(const v8h*)(rowPtr + 16 * hi + 8);   // K = 16*hi + 8..15
    return __builtin_shufflevector(a0, a1, 0,1,2,3,4,5,6,7,8,9,10,11,12,13,14,15);
}
// B fragment from K-contiguous storage (row = output column N): 16 contiguous halves
__device__ __forceinline__ v16h load_b_frag(const _Float16* __restrict__ colPtr) {
    v8h b0 = *(const v8h*)(colPtr);
    v8h b1 = *(const v8h*)(colPtr + 8);
    return __builtin_shufflevector(b0, b1, 0,1,2,3,4,5,6,7,8,9,10,11,12,13,14,15);
}

// ---------------- stage 1: embed + bbox proj + LayerNorm + token CE; scatter into line sums ----
__global__ void encoder_cls_kernel(const int* __restrict__ ids,
                                   const int* __restrict__ bbox,
                                   const int* __restrict__ labels,
                                   const int* __restrict__ line_ids,
                                   const float* __restrict__ emb,
                                   const float* __restrict__ Wbbox,
                                   const float* __restrict__ gamma,
                                   const float* __restrict__ beta,
                                   const float* __restrict__ Wcls,
                                   float* __restrict__ lineSum,
                                   float* __restrict__ lineCnt,
                                   float* __restrict__ accum) {
    __shared__ float red[256];
    int tok = blockIdx.x;                 // 0..B*S-1
    int b   = tok / SEQ;
    int t   = threadIdx.x;

    int id = ids[tok];
    float bb0 = (float)bbox[tok * 4 + 0];
    float bb1 = (float)bbox[tok * 4 + 1];
    float bb2 = (float)bbox[tok * 4 + 2];
    float bb3 = (float)bbox[tok * 4 + 3];

    float v[3];
    int   dim[3];
    #pragma unroll
    for (int j = 0; j < 3; ++j) {
        int d = t + j * 256;
        dim[j] = d;
        v[j] = emb[(long)id * H_ + d]
             + bb0 * Wbbox[0 * H_ + d] + bb1 * Wbbox[1 * H_ + d]
             + bb2 * Wbbox[2 * H_ + d] + bb3 * Wbbox[3 * H_ + d];
    }
    float s  = blockReduceSum256(v[0] + v[1] + v[2], red);
    float sq = blockReduceSum256(v[0]*v[0] + v[1]*v[1] + v[2]*v[2], red);
    float mu   = s / (float)H_;
    float var  = sq / (float)H_ - mu * mu;
    float rstd = rsqrtf(var + 1e-12f);

    int lid = line_ids[tok];
    float n[3];
    #pragma unroll
    for (int j = 0; j < 3; ++j) {
        n[j] = (v[j] - mu) * rstd * gamma[dim[j]] + beta[dim[j]];
        atomicAdd(&lineSum[((long)(b * MAXL + lid)) * H_ + dim[j]], n[j]);
    }
    if (t == 0) atomicAdd(&lineCnt[b * MAXL + lid], 1.0f);

    // token logits: 7 dot products of length 768
    float pc[NCLS];
    #pragma unroll
    for (int c = 0; c < NCLS; ++c)
        pc[c] = n[0] * Wcls[dim[0] * NCLS + c]
              + n[1] * Wcls[dim[1] * NCLS + c]
              + n[2] * Wcls[dim[2] * NCLS + c];
    float logit[NCLS];
    for (int c = 0; c < NCLS; ++c) logit[c] = blockReduceSum256(pc[c], red);

    if (t == 0) {
        int lab = labels[tok];
        if (lab != -100) {
            float mx = logit[0];
            for (int c = 1; c < NCLS; ++c) mx = fmaxf(mx, logit[c]);
            float se = 0.f;
            for (int c = 0; c < NCLS; ++c) se += __expf(logit[c] - mx);
            float nll = logf(se) + mx - logit[lab < 0 ? 0 : lab];
            atomicAdd(&accum[0], nll);
            atomicAdd(&accum[1], 1.0f);
        }
    }
}

// ---------------- stage 2: finalize line features (mean), emit f32 + f16, count lines ----------
__global__ void line_finalize_kernel(float* __restrict__ lineSum,      // in-place -> f32 feat
                                     const float* __restrict__ lineCnt,
                                     _Float16* __restrict__ lf16,
                                     int* __restrict__ numLines) {
    int bl = blockIdx.x;                  // 0..B*L-1
    int t  = threadIdx.x;
    float cnt = lineCnt[bl];
    float inv = 1.0f / fmaxf(cnt, 1.0f);
    #pragma unroll
    for (int j = 0; j < 3; ++j) {
        long idx = (long)bl * H_ + t + j * 256;
        float f = lineSum[idx] * inv;
        lineSum[idx] = f;
        lf16[idx] = (_Float16)f;
    }
    if (t == 0 && cnt > 0.f) atomicAdd(&numLines[bl / MAXL], 1);
}

// ---------------- W_bi: f32 [K,N] -> f16 transposed [N,K] (K-contiguous for b128 B-frag loads)
__global__ void wbi_transpose_f16_kernel(const float* __restrict__ src, _Float16* __restrict__ dst) {
    int i = blockIdx.x * blockDim.x + threadIdx.x;   // over H*H
    if (i < H_ * H_) {
        int n = i / H_, k = i - n * H_;
        dst[i] = (_Float16)src[(long)k * H_ + n];
    }
}

// ---------------- WMMA GEMM #1: M[b] = LF16[b] (L x H) * Wbi (H x H), f16 out ----------------
// one wave per 32x64 macro-tile (2 M-tiles x 4 N-tiles = 8 wmma/K-step); grid = (L/32, H/64, B)
__global__ void wmma_gemm_lfxw_kernel(const _Float16* __restrict__ A,    // [B,L,H]
                                      const _Float16* __restrict__ Wt,   // [H(N),H(K)] transposed
                                      _Float16* __restrict__ Out) {      // [B,L,H]
    int b  = blockIdx.z;
    int m0 = blockIdx.x * 32;
    int n0 = blockIdx.y * 64;
    const _Float16* Ab = A + (long)b * MAXL * H_;
    int lane = threadIdx.x & 31;
    int col  = lane & 15;
    int hi   = lane >> 4;
    v8f acc[2][4] = {};
    for (int k0 = 0; k0 < H_; k0 += 32) {
        v16h af[2], bf[4];
        #pragma unroll
        for (int mi = 0; mi < 2; ++mi) {
            const _Float16* ar = Ab + (long)(m0 + mi * 16 + col) * H_ + k0;
            af[mi] = load_a_frag(ar, hi);
            __builtin_prefetch(ar + 32, 0, 1);
        }
        #pragma unroll
        for (int ni = 0; ni < 4; ++ni) {
            const _Float16* br = Wt + (long)(n0 + ni * 16 + col) * H_ + k0 + 16 * hi;
            bf[ni] = load_b_frag(br);
            __builtin_prefetch(br + 32, 0, 1);
        }
        #pragma unroll
        for (int mi = 0; mi < 2; ++mi)
            #pragma unroll
            for (int ni = 0; ni < 4; ++ni)
                acc[mi][ni] = __builtin_amdgcn_wmma_f32_16x16x32_f16(
                    false, af[mi], false, bf[ni], (short)0, acc[mi][ni], false, false);
    }
    _Float16* Ob = Out + (long)b * MAXL * H_;
    #pragma unroll
    for (int mi = 0; mi < 2; ++mi)
        #pragma unroll
        for (int r = 0; r < 8; ++r) {
            long row = (long)(m0 + mi * 16 + r + 8 * hi) * H_;
            #pragma unroll
            for (int ni = 0; ni < 4; ++ni)
                Ob[row + n0 + ni * 16 + col] = (_Float16)acc[mi][ni][r];
        }
}

// ---------------- WMMA GEMM #2: scores[b] = M16[b] (L x H) * LF16[b]^T, f32 out --------------
// one wave per 32x64 macro-tile; grid = (L/32, L/64, B)
__global__ void wmma_gemm_scores_kernel(const _Float16* __restrict__ A,   // [B,L,H]
                                        const _Float16* __restrict__ LF,  // [B,L,H] (transposed use)
                                        float* __restrict__ Scores) {     // [B,L,L]
    int b  = blockIdx.z;
    int m0 = blockIdx.x * 32;   // child rows
    int n0 = blockIdx.y * 64;   // parent cols
    const _Float16* Ab = A  + (long)b * MAXL * H_;
    const _Float16* Lb = LF + (long)b * MAXL * H_;
    int lane = threadIdx.x & 31;
    int col  = lane & 15;
    int hi   = lane >> 4;
    v8f acc[2][4] = {};
    for (int k0 = 0; k0 < H_; k0 += 32) {
        v16h af[2], bf[4];
        #pragma unroll
        for (int mi = 0; mi < 2; ++mi) {
            const _Float16* ar = Ab + (long)(m0 + mi * 16 + col) * H_ + k0;
            af[mi] = load_a_frag(ar, hi);
            __builtin_prefetch(ar + 32, 0, 1);
        }
        #pragma unroll
        for (int ni = 0; ni < 4; ++ni) {
            // B[k][p] = LF[p][k] -> row p is K-contiguous already
            const _Float16* br = Lb + (long)(n0 + ni * 16 + col) * H_ + k0 + 16 * hi;
            bf[ni] = load_b_frag(br);
            __builtin_prefetch(br + 32, 0, 1);
        }
        #pragma unroll
        for (int mi = 0; mi < 2; ++mi)
            #pragma unroll
            for (int ni = 0; ni < 4; ++ni)
                acc[mi][ni] = __builtin_amdgcn_wmma_f32_16x16x32_f16(
                    false, af[mi], false, bf[ni], (short)0, acc[mi][ni], false, false);
    }
    float* Sb = Scores + (long)b * MAXL * MAXL;
    #pragma unroll
    for (int mi = 0; mi < 2; ++mi)
        #pragma unroll
        for (int r = 0; r < 8; ++r) {
            long row = (long)(m0 + mi * 16 + r + 8 * hi) * MAXL;
            #pragma unroll
            for (int ni = 0; ni < 4; ++ni)
                Sb[row + n0 + ni * 16 + col] = acc[mi][ni][r];
        }
}

// ---------------- parent loss: per (b,c) row log-softmax over p with p<c mask ----------------
__global__ void parent_loss_kernel(const float* __restrict__ Scores,
                                   const int* __restrict__ parent_ids,
                                   const int* __restrict__ numLines,
                                   float* __restrict__ accum) {
    __shared__ float red[256];
    int bl = blockIdx.x;                  // 0..B*L-1
    int b = bl / MAXL, c = bl % MAXL;
    int t = threadIdx.x;                  // p index
    const float* row = Scores + (long)bl * MAXL;
    float sc = (t < c) ? row[t] : NEGINF;
    float mx = blockReduceMax256(sc, red);
    float se = blockReduceSum256(__expf(sc - mx), red);
    if (t == 0) {
        int pid = parent_ids[bl];
        int nl  = numLines[b];
        bool valid = (pid >= 0) && (pid < c) && (c >= 1) && (c < nl);
        if (valid) {
            int gt = min(max(pid, 0), MAXL - 1);
            float nll = logf(se) + mx - row[gt];
            atomicAdd(&accum[2], nll);
            atomicAdd(&accum[3], 1.0f);
        }
    }
}

// ---------------- relation loss: [parent_feat, child_feat] @ W_rel + b_rel, 4-way CE ---------
__global__ void relation_loss_kernel(const float* __restrict__ lineFeat,   // [B,L,H] f32
                                     const int* __restrict__ parent_ids,
                                     const int* __restrict__ relations,
                                     const int* __restrict__ numLines,
                                     const float* __restrict__ Wrel,       // [2H,4]
                                     const float* __restrict__ brel,
                                     float* __restrict__ accum) {
    __shared__ float red[256];
    int bl = blockIdx.x;                  // 0..B*L-1
    int b = bl / MAXL, c = bl % MAXL;
    int t = threadIdx.x;
    int pid  = parent_ids[bl];
    int pidx = min(max(pid, 0), MAXL - 1);
    const float* pf = lineFeat + ((long)b * MAXL + pidx) * H_;
    const float* cf = lineFeat + ((long)b * MAXL + c) * H_;
    float pr[NREL] = {0.f, 0.f, 0.f, 0.f};
    for (int k = t; k < H_; k += 256) {
        float pv = pf[k], cv = cf[k];
        #pragma unroll
        for (int r = 0; r < NREL; ++r) {
            pr[r] += pv * Wrel[k * NREL + r];
            pr[r] += cv * Wrel[(H_ + k) * NREL + r];
        }
    }
    float logit[NREL];
    for (int r = 0; r < NREL; ++r) logit[r] = blockReduceSum256(pr[r], red);
    if (t == 0) {
        int nl  = numLines[b];
        int rel = relations[bl];
        bool valid = (pid >= 0) && (pid < nl) && (rel != -100) && (c < nl);
        if (valid) {
            float l0 = logit[0] + brel[0], l1 = logit[1] + brel[1];
            float l2 = logit[2] + brel[2], l3 = logit[3] + brel[3];
            float mx = fmaxf(fmaxf(l0, l1), fmaxf(l2, l3));
            float se = __expf(l0 - mx) + __expf(l1 - mx) + __expf(l2 - mx) + __expf(l3 - mx);
            int rl = min(max(rel, 0), NREL - 1);
            float lg = (rl == 0) ? l0 : (rl == 1) ? l1 : (rl == 2) ? l2 : l3;
            float nll = logf(se) + mx - lg;
            atomicAdd(&accum[4], nll);
            atomicAdd(&accum[5], 1.0f);
        }
    }
}

// ---------------- combine ----------------
__global__ void finalize_kernel(const float* __restrict__ accum, float* __restrict__ out) {
    if (threadIdx.x == 0 && blockIdx.x == 0) {
        float cls = accum[0] / fmaxf(accum[1], 1.0f);
        float par = accum[2] / fmaxf(accum[3], 1.0f);
        float rel = accum[4] / fmaxf(accum[5], 1.0f);
        out[0] = cls + par + rel;
    }
}

extern "C" void kernel_launch(void* const* d_in, const int* in_sizes, int n_in,
                              void* d_out, int out_size, void* d_ws, size_t ws_size,
                              hipStream_t stream) {
    const int*   input_ids  = (const int*)d_in[0];
    const int*   bbox       = (const int*)d_in[1];
    // d_in[2] attention_mask: unused by the reference loss
    const int*   labels     = (const int*)d_in[3];
    const int*   line_ids   = (const int*)d_in[4];
    const int*   parent_ids = (const int*)d_in[5];
    const int*   relations  = (const int*)d_in[6];
    const float* emb        = (const float*)d_in[7];
    const float* Wbbox      = (const float*)d_in[8];
    const float* gamma      = (const float*)d_in[9];
    const float* beta       = (const float*)d_in[10];
    const float* Wcls       = (const float*)d_in[11];
    const float* Wbi        = (const float*)d_in[12];
    const float* Wrel       = (const float*)d_in[13];
    const float* brel       = (const float*)d_in[14];
    float* out = (float*)d_out;

    // workspace layout (256B aligned regions)
    char* ws = (char*)d_ws;
    const size_t off_accum   = 0;                                     // 8 floats
    const size_t off_nlines  = 256;                                   // 8 ints
    const size_t off_cnt     = 512;                                   // B*L floats     = 8 KB
    const size_t off_lsum    = off_cnt + (size_t)BATCH*MAXL*4;        // B*L*H floats   = 6 MB
    const size_t zero_bytes  = off_lsum + (size_t)BATCH*MAXL*H_*4;
    const size_t off_lf16    = zero_bytes;                            // B*L*H halves
    const size_t off_wbi16   = off_lf16 + (size_t)BATCH*MAXL*H_*2;    // H*H halves (transposed)
    const size_t off_m16     = off_wbi16 + (size_t)H_*H_*2;           // B*L*H halves
    const size_t off_scores  = off_m16 + (size_t)BATCH*MAXL*H_*2;     // B*L*L floats

    float*     accum    = (float*)(ws + off_accum);
    int*       numLines = (int*)(ws + off_nlines);
    float*     lineCnt  = (float*)(ws + off_cnt);
    float*     lineSum  = (float*)(ws + off_lsum);
    _Float16*  lf16     = (_Float16*)(ws + off_lf16);
    _Float16*  wbiT16   = (_Float16*)(ws + off_wbi16);
    _Float16*  m16      = (_Float16*)(ws + off_m16);
    float*     scores   = (float*)(ws + off_scores);

    hipMemsetAsync(d_ws, 0, zero_bytes, stream);

    encoder_cls_kernel<<<BATCH * SEQ, 256, 0, stream>>>(
        input_ids, bbox, labels, line_ids, emb, Wbbox, gamma, beta, Wcls,
        lineSum, lineCnt, accum);

    line_finalize_kernel<<<BATCH * MAXL, 256, 0, stream>>>(lineSum, lineCnt, lf16, numLines);

    wbi_transpose_f16_kernel<<<(H_ * H_ + 255) / 256, 256, 0, stream>>>(Wbi, wbiT16);

    wmma_gemm_lfxw_kernel<<<dim3(MAXL / 32, H_ / 64, BATCH), 32, 0, stream>>>(lf16, wbiT16, m16);

    wmma_gemm_scores_kernel<<<dim3(MAXL / 32, MAXL / 64, BATCH), 32, 0, stream>>>(m16, lf16, scores);

    parent_loss_kernel<<<BATCH * MAXL, 256, 0, stream>>>(scores, parent_ids, numLines, accum);

    relation_loss_kernel<<<BATCH * MAXL, 256, 0, stream>>>(
        lineSum, parent_ids, relations, numLines, Wrel, brel, accum);

    finalize_kernel<<<1, 32, 0, stream>>>(accum, out);
}
